// paper_bi_36344013258826
// MI455X (gfx1250) — compile-verified, matrix operand order checked by reading, not run on hardware
//
#include <hip/hip_runtime.h>
#include <hip/hip_bf16.h>
#include <math.h>

// ---------------------------------------------------------------------------
// FFT long-convolution (bidirectional Hyena filter) for MI455X / gfx1250.
//   y[b,h,:] = irfft( rfft(x,2L) * rfft(k_bi,2L) )[:L] + x * D[h]
// B=8 H=1024 L=4096, N=2L=8192 = 16*16*32 Cooley-Tukey.
// Each DFT stage runs on the WMMA pipes via the stacked-real trick:
//   Re = [Fr|-Fi]x[Br;Bi],  Im = [Fi|Fr]x[Br;Bi]  -> v_wmma_f32_16x16x32_f16.
// Both real inputs (x and k_bi) are packed into ONE complex FFT (z = x + i*k).
// Inputs are staged into LDS by the Tensor Data Mover (tensor_load_to_lds,
// TENSORcnt), not by per-lane VMEM loads.
// Memory-bound: ~290 MB total HBM traffic -> ~12.5 us at 23.3 TB/s.
// ---------------------------------------------------------------------------

typedef __attribute__((ext_vector_type(16))) _Float16     v16h;
typedef __attribute__((ext_vector_type(8)))  float        v8f;
typedef __attribute__((ext_vector_type(4)))  unsigned int v4u;
typedef __attribute__((ext_vector_type(8)))  int          v8i;
typedef __attribute__((ext_vector_type(4)))  int          v4i;

#define WMMA_F16(A, B, C) \
  __builtin_amdgcn_wmma_f32_16x16x32_f16(false, (A), false, (B), (short)0, (C), false, false)

constexpr int   Bb     = 8;
constexpr int   Hh     = 1024;
constexpr int   Ll     = 4096;
constexpr int   Nf     = 8192;            // FFT length = 2*L = 16*16*32
constexpr float TWO_PI = 6.28318530717958647692f;

// position p = d0 + 16*d1 + 256*d2  (d0,d1 in [0,16), d2 in [0,32))
// frequency stored at p is  f(p) = d2 + 32*d1 + 512*d0  (an involution pair).
__device__ __forceinline__ int posOfFreq(int f) {
  // f = f3 + 32*f2 + 512*f1  ->  p = f1 + 16*f2 + 256*f3
  return (f >> 9) + (((f >> 5) & 15) << 4) + ((f & 31) << 8);
}

template <int AXIS>
__device__ __forceinline__ int posAx(int r, int c) {
  if constexpr (AXIS == 0) return r + (c << 4);                               // c = d1 + 16*d2
  else if constexpr (AXIS == 1) return (c & 15) + (r << 4) + ((c >> 4) << 8); // c = d0 + 16*d2
  else return c + (r << 8);                                                   // c = d0 + 16*d1
}

// K index of element e in the wave32 16-bit A/B fragment (v16h), per the
// documented 16x32 layout: lanes 0-15 cover K {0..7,16..23}, lanes 16-31
// cover K {8..15,24..31}; within a lane, vector elems walk K pairs.
__device__ __forceinline__ int fragK(int e, int half) {
  return (e < 8 ? e : e + 8) + 8 * half;
}

// ----- Tensor Data Mover: DMA one contiguous row (nelem f32) into LDS -------
// D# built per ISA 8.3/8.4: group0 {count=1, lds_addr, global_addr, type=2},
// group1 {data_size=4B, tensor_dim0=tile_dim0=nelem, tensor_dim1=tile_dim1=1,
// tensor_dim0_stride=nelem}. Groups 2/3 zero (<=2D tensor).
// This toolchain's builtin takes 6 args (clang-23 lane):
//   (uint32x4 g0, int32x8 g1, int32x4, int32x4, int32x8, i32 cpol)
#if defined(__gfx1250__) && __has_builtin(__builtin_amdgcn_tensor_load_to_lds)
#define HAVE_TDM 1
__device__ __forceinline__ void tdm_load_row(const float* gsrc, float* ldst, int nelem) {
  const unsigned long long ga  = (unsigned long long)(size_t)gsrc;
  const unsigned int       lds = (unsigned int)(size_t)ldst;

  v4u g0;
  g0[0] = 1u;                                                    // count=1, user D#
  g0[1] = lds;                                                   // lds_addr [63:32]
  g0[2] = (unsigned int)(ga & 0xffffffffu);                      // global_addr lo
  g0[3] = (unsigned int)((ga >> 32) & 0x01ffffffu) | (2u << 30); // ga hi | type=2

  v8i g1;
  g1[0] = (2 << 16);                                   // data_size=2 (4B), mask=0
  g1[1] = (nelem & 0xffff) << 16;                      // [63:48]  tensor_dim0 lo16
  g1[2] = ((nelem >> 16) & 0xffff) | (1 << 16);        // dim0 hi16 | tensor_dim1=1 lo
  g1[3] = (nelem & 0xffff) << 16;                      // dim1 hi=0 | tile_dim0
  g1[4] = 1;                                           // tile_dim1=1, tile_dim2=0
  g1[5] = nelem;                                       // tensor_dim0_stride lo32
  g1[6] = 0;                                           // stride hi | dim1_stride lo
  g1[7] = 0;

  const v4i zz4 = {0, 0, 0, 0};
  const v8i zz8 = {0, 0, 0, 0, 0, 0, 0, 0};
  __builtin_amdgcn_tensor_load_to_lds(g0, g1, zz4, zz4, zz8, 0);
}
#else
#define HAVE_TDM 0
#endif

// ----- radix-16 DFT along one position digit, on the WMMA pipe --------------
// sgn = -1 forward (W^-), +1 inverse (W^+).
template <int AXIS>
__device__ void dft16_stage(float* __restrict__ zre, float* __restrict__ zim,
                            float sgn, int lane, int wave) {
  const int half = lane >> 4;
  const int nl   = lane & 15;
  const int M    = nl;  // A-matrix row held by this lane

  // Constant DFT-16 operand fragments: Are = [Fr | -Fi], Aim = [Fi | Fr]
  v16h aRe, aIm;
#pragma unroll
  for (int e = 0; e < 16; ++e) {
    const int   K  = fragK(e, half);
    const int   cc = K & 15;
    const float th = TWO_PI * (float)((M * cc) & 15) * (1.0f / 16.0f);
    const float fr = __cosf(th);
    const float fi = sgn * __sinf(th);
    aRe[e] = (_Float16)(K < 16 ? fr : -fi);
    aIm[e] = (_Float16)(K < 16 ? fi : fr);
  }

  // 512 columns -> 32 tiles of 16; each of the 8 waves owns 4 tiles.
#pragma unroll
  for (int tt = 0; tt < 4; ++tt) {
    const int t = wave + tt * 8;
    const int c = t * 16 + nl;

    v16h bF;  // stacked [Re rows 0..15 ; Im rows 0..15]
#pragma unroll
    for (int e = 0; e < 16; ++e) {
      const int K = fragK(e, half);
      const int r = K & 15;
      const int p = posAx<AXIS>(r, c);
      bF[e] = (_Float16)(K < 16 ? zre[p] : zim[p]);
    }

    v8f z0 = {0.f, 0.f, 0.f, 0.f, 0.f, 0.f, 0.f, 0.f};
    v8f dRe = WMMA_F16(aRe, bF, z0);
    v8f dIm = WMMA_F16(aIm, bF, z0);

#pragma unroll
    for (int r = 0; r < 8; ++r) {
      const int row = r + 8 * half;  // documented C/D layout
      const int p   = posAx<AXIS>(row, c);
      zre[p] = dRe[r];
      zim[p] = dIm[r];
    }
  }
}

// ----- radix-32 DFT along the high digit (2x2 blocked 16x16 WMMAs) ----------
__device__ void dft32_stage(float* __restrict__ zre, float* __restrict__ zim,
                            float sgn, int lane, int wave) {
  const int half = lane >> 4;
  const int nl   = lane & 15;
  const int M    = nl;

  v16h aRe[2][2], aIm[2][2];  // [row-block][col-block]
#pragma unroll
  for (int rh = 0; rh < 2; ++rh) {
#pragma unroll
    for (int cb = 0; cb < 2; ++cb) {
      const int row = rh * 16 + M;
#pragma unroll
      for (int e = 0; e < 16; ++e) {
        const int   K    = fragK(e, half);
        const int   colF = cb * 16 + (K & 15);
        const float th   = TWO_PI * (float)((row * colF) & 31) * (1.0f / 32.0f);
        const float fr   = __cosf(th);
        const float fi   = sgn * __sinf(th);
        aRe[rh][cb][e] = (_Float16)(K < 16 ? fr : -fi);
        aIm[rh][cb][e] = (_Float16)(K < 16 ? fi : fr);
      }
    }
  }

  // 256 columns -> 16 tiles of 16; each wave owns 2 tiles.
#pragma unroll
  for (int tt = 0; tt < 2; ++tt) {
    const int t = wave + tt * 8;
    const int c = t * 16 + nl;

    v16h b0, b1;  // data rows 0..15 and 16..31, each stacked [Re;Im]
#pragma unroll
    for (int e = 0; e < 16; ++e) {
      const int K = fragK(e, half);
      const int r = K & 15;
      b0[e] = (_Float16)(K < 16 ? zre[posAx<2>(r, c)] : zim[posAx<2>(r, c)]);
      b1[e] = (_Float16)(K < 16 ? zre[posAx<2>(r + 16, c)] : zim[posAx<2>(r + 16, c)]);
    }

    v8f outRe[2], outIm[2];
#pragma unroll
    for (int rh = 0; rh < 2; ++rh) {
      v8f acc = {0.f, 0.f, 0.f, 0.f, 0.f, 0.f, 0.f, 0.f};
      acc = WMMA_F16(aRe[rh][0], b0, acc);   // chained accumulation over col-blocks
      acc = WMMA_F16(aRe[rh][1], b1, acc);
      outRe[rh] = acc;
      v8f aci = {0.f, 0.f, 0.f, 0.f, 0.f, 0.f, 0.f, 0.f};
      aci = WMMA_F16(aIm[rh][0], b0, aci);
      aci = WMMA_F16(aIm[rh][1], b1, aci);
      outIm[rh] = aci;
    }

#pragma unroll
    for (int rh = 0; rh < 2; ++rh) {
#pragma unroll
      for (int r = 0; r < 8; ++r) {
        const int row = rh * 16 + r + 8 * half;
        const int p   = posAx<2>(row, c);
        zre[p] = outRe[rh][r];
        zim[p] = outIm[rh][r];
      }
    }
  }
}

// ----- inter-stage twiddle passes (elementwise, VALU/TRANS) -----------------
// which==0:  W512^{sgn*d1*d2}        which==1:  W8192^{sgn*d0*(d2+32*d1)}
__device__ void twiddle_pass(float* __restrict__ zre, float* __restrict__ zim,
                             float sgn, int which, int tid) {
  for (int p = tid; p < Nf; p += 256) {
    const int d0 = p & 15, d1 = (p >> 4) & 15, d2 = p >> 8;
    float ang;
    if (which == 0)
      ang = sgn * TWO_PI * (float)((d1 * d2) & 511) * (1.0f / 512.0f);
    else
      ang = sgn * TWO_PI * (float)((d0 * (d2 + (d1 << 5))) & 8191) * (1.0f / 8192.0f);
    const float c = __cosf(ang), s = __sinf(ang);
    const float zr = zre[p], zi = zim[p];
    zre[p] = zr * c - zi * s;
    zim[p] = zr * s + zi * c;
  }
}

// ---------------------------------------------------------------------------
__global__ void __launch_bounds__(256)
fftconv_wmma_kernel(const float* __restrict__ x, const float* __restrict__ k,
                    const float* __restrict__ Dw, float* __restrict__ y) {
  extern __shared__ float smem[];
  float* zre = smem;            // 8192 floats
  float* zim = smem + Nf;       // 8192 floats
  float* kst = smem + 2 * Nf;   // 4096 floats: k1 staging (80 KB LDS total)

  const int tid  = threadIdx.x;
  const int lane = tid & 31;
  const int wave = tid >> 5;
  const int bid  = blockIdx.x;
  const int h    = bid & (Hh - 1);
  const int b    = bid >> 10;

  const float* xrow = x + ((size_t)b * Hh + h) * Ll;
  const float* k0   = k + (size_t)h * Ll;           // causal half
  const float* k1   = k + (size_t)(Hh + h) * Ll;    // anticausal half

#if HAVE_TDM
  // ---- TDM staging: x -> zre[0:L], k0 -> zim[0:L], k1 -> kst (forward).
  // Wave-level DMA (EXEC ignored); one wave issues, TENSORcnt gates.
  if (wave == 0) {
    tdm_load_row(xrow, zre, Ll);
    tdm_load_row(k0, zim, Ll);
    tdm_load_row(k1, kst, Ll);
    __builtin_amdgcn_s_wait_tensorcnt(0);
  }
  // Keep the x row warm for the epilogue D-term re-read.
  for (int off = tid * 128; off < Ll; off += 256 * 128)
    __builtin_prefetch(xrow + off, 0, 1);
  __syncthreads();
  // Zero-pad x half, reverse k1 into upper half: z = x_pad + i * k_bi.
  for (int j = tid; j < Ll; j += 256) {
    zre[Ll + j] = 0.0f;
    zim[Ll + j] = kst[Ll - 1 - j];   // k_bi[L+t] = k1[L-1-t]
  }
#else
  // Fallback: direct VMEM -> LDS packing.
  for (int j = tid; j < Nf; j += 256) {
    zre[j] = (j < Ll) ? xrow[j] : 0.0f;
    zim[j] = (j < Ll) ? k0[j] : k1[Nf - 1 - j];
  }
#endif
  __syncthreads();

  // ---- forward FFT (W^-): radix-32 axis2 -> tw -> r16 axis1 -> tw -> r16 axis0
  dft32_stage(zre, zim, -1.0f, lane, wave);
  __syncthreads();
  twiddle_pass(zre, zim, -1.0f, 0, tid);
  __syncthreads();
  dft16_stage<1>(zre, zim, -1.0f, lane, wave);
  __syncthreads();
  twiddle_pass(zre, zim, -1.0f, 1, tid);
  __syncthreads();
  dft16_stage<0>(zre, zim, -1.0f, lane, wave);
  __syncthreads();

  // ---- pointwise: unpack the two real spectra from Z, multiply, fold 1/N.
  // Spectrum for frequency f lives at posOfFreq(f) (mixed-radix digit reversal).
  const float invN = 1.0f / (float)Nf;
  for (int f = tid; f <= Nf / 2; f += 256) {
    const int p  = posOfFreq(f);
    const int fm = (Nf - f) & (Nf - 1);
    const int q  = posOfFreq(fm);
    const float zr1 = zre[p], zi1 = zim[p];
    const float zr2 = zre[q], zi2 = zim[q];
    // X[f] = (Z[f]+conj(Z[-f]))/2 ; K[f] = (Z[f]-conj(Z[-f]))/(2i)
    const float xr = 0.5f * (zr1 + zr2), xi = 0.5f * (zi1 - zi2);
    const float kr = 0.5f * (zi1 + zi2), ki = 0.5f * (zr2 - zr1);
    const float yr = (xr * kr - xi * ki) * invN;
    const float yi = (xr * ki + xi * kr) * invN;
    zre[p] = yr;
    zim[p] = yi;
    if (q != p) { zre[q] = yr; zim[q] = -yi; }  // Y[-f] = conj(Y[f])
  }
  __syncthreads();

  // ---- inverse FFT (W^+): mirrored stage order, ends in natural order.
  dft16_stage<0>(zre, zim, +1.0f, lane, wave);
  __syncthreads();
  twiddle_pass(zre, zim, +1.0f, 1, tid);
  __syncthreads();
  dft16_stage<1>(zre, zim, +1.0f, lane, wave);
  __syncthreads();
  twiddle_pass(zre, zim, +1.0f, 0, tid);
  __syncthreads();
  dft32_stage(zre, zim, +1.0f, lane, wave);
  __syncthreads();

  // ---- epilogue: y = Re(ifft) + x * D[h]   (float4 -> b128 traffic)
  const float Dh = Dw[h];
  float* yrow    = y + ((size_t)b * Hh + h) * Ll;
  for (int j = tid * 4; j < Ll; j += 256 * 4) {
    const float4 xv = *reinterpret_cast<const float4*>(xrow + j);
    float4 o;
    o.x = zre[j + 0] + xv.x * Dh;
    o.y = zre[j + 1] + xv.y * Dh;
    o.z = zre[j + 2] + xv.z * Dh;
    o.w = zre[j + 3] + xv.w * Dh;
    *reinterpret_cast<float4*>(yrow + j) = o;
  }
}

// ---------------------------------------------------------------------------
extern "C" void kernel_launch(void* const* d_in, const int* in_sizes, int n_in,
                              void* d_out, int out_size, void* d_ws, size_t ws_size,
                              hipStream_t stream) {
  const float* x  = (const float*)d_in[0];  // (B, H, L) f32
  const float* k  = (const float*)d_in[1];  // (2, H, L) f32
  const float* Dw = (const float*)d_in[2];  // (1, H, 1) f32
  float*       y  = (float*)d_out;          // (B, H, L) f32

  const size_t shmem = (2 * (size_t)Nf + Ll) * sizeof(float);  // 80 KB dynamic LDS
  (void)hipFuncSetAttribute((const void*)fftconv_wmma_kernel,
                            hipFuncAttributeMaxDynamicSharedMemorySize, (int)shmem);

  fftconv_wmma_kernel<<<dim3(Bb * Hh), dim3(256), shmem, stream>>>(x, k, Dw, y);
}
